// S4Kernel_24240795419332
// MI455X (gfx1250) — compile-verified, matrix operand order checked by reading, not run on hardware
//
#include <hip/hip_runtime.h>

// ---------------------------------------------------------------------------
// S4 / selective-SSM forward for MI455X (gfx1250, wave32, WMMA).
//   B=4, L=4096, D_MODEL=1024, STATE=16  ->  16384 tokens.
// Big cost: delta GEMM (16384 x 1024 x 1024, 34.4 GFLOP) -> bf16 WMMA, f32 acc,
// async global->LDS staging (ASYNCcnt) with double-buffered LDS.
// ---------------------------------------------------------------------------

#define NBATCH   4
#define SEQLEN   4096
#define DM       1024
#define ST       16
#define NTOK     (NBATCH * SEQLEN)          // 16384

typedef __attribute__((ext_vector_type(16))) __bf16 v16bf;
typedef __attribute__((ext_vector_type(8)))  __bf16 v8bf;
typedef __attribute__((ext_vector_type(8)))  float  v8f;

__device__ __forceinline__ float softplusf(float v) {
    return (v > 20.0f) ? v : log1pf(__expf(v));
}

// CDNA5 async copy: global memory -> LDS, tracked by ASYNCcnt (ISA 10.x / 15.18.3).
// lds_off = low 32 bits of the generic pointer (LDS aperture puts the byte
// address in addr[31:0]); gptr = 64-bit global address in a VGPR pair.
__device__ __forceinline__ void async_load_b128(unsigned lds_off, const void* gptr) {
    asm volatile("global_load_async_to_lds_b128 %0, %1, off"
                 :: "v"(lds_off), "v"(gptr) : "memory");
}
#define WAIT_ASYNCCNT_3() asm volatile("s_wait_asynccnt 3" ::: "memory")
#define WAIT_ASYNCCNT_0() asm volatile("s_wait_asynccnt 0" ::: "memory")

// ---------------------------------------------------------------------------
// fp32 -> bf16 bulk convert (vectorized, grid-stride; n8 = elements/8)
// ---------------------------------------------------------------------------
__global__ void cvt_bf16_kernel(const float* __restrict__ src,
                                __bf16* __restrict__ dst, int n8) {
    int i = blockIdx.x * blockDim.x + threadIdx.x;
    const int stride = gridDim.x * blockDim.x;
    for (; i < n8; i += stride) {
        const float4 a = ((const float4*)src)[2 * i];
        const float4 b = ((const float4*)src)[2 * i + 1];
        v8bf o = { (__bf16)a.x, (__bf16)a.y, (__bf16)a.z, (__bf16)a.w,
                   (__bf16)b.x, (__bf16)b.y, (__bf16)b.z, (__bf16)b.w };
        ((v8bf*)dst)[i] = o;
    }
}

// ---------------------------------------------------------------------------
// delta GEMM:  logits[t,e] = sum_d Xbf[t,d]*Wbf[e,d] + b_delta[e]
// fused softplus + row-sum over e (atomicAdd into delta_sum[t]; /1024 later).
//
// Workgroup: 256 threads = 8 waves; tile M=128 (tokens) x N=64 (e).
// Wave w: mi = w&3 (32-row strip), ni = w>>2 (32-col strip) -> 4 accumulators
// (2 A-frags x 2 B-frags, operand reuse). K loop: 32 steps of K=32.
// LDS: double-buffered A(128x32) + B(64x32, N-major) bf16 tiles, filled by
// GLOBAL_LOAD_ASYNC_TO_LDS_B128 (3 per thread per stage), s_wait_asynccnt
// pipelined one stage ahead.
//
// Fragment layout (CDNA5 ISA, 16-bit A 16x32): lane m = lane&15, kh = lane>>4,
//   element e -> k = (e&7) + 8*kh + 16*(e>>3)   (two contiguous 16B LDS loads).
// B fragment mirrored (n = lane&15, same k index function).
// ---------------------------------------------------------------------------
__global__ void __launch_bounds__(256)
delta_gemm_kernel(const __bf16* __restrict__ Xbf,
                  const __bf16* __restrict__ Wbf,
                  const float* __restrict__ b_delta,
                  float* __restrict__ delta_sum) {
    constexpr int LDT = 48;                      // padded row stride (96B)
    constexpr int STAGE_ELEMS = (128 + 64) * LDT;
    constexpr unsigned STAGE_BYTES = STAGE_ELEMS * 2;      // 18432
    constexpr unsigned B_OFF_BYTES = 128 * LDT * 2;        // 12288
    __shared__ __bf16 smem[2 * STAGE_ELEMS];               // 36 KB

    const int tid  = threadIdx.x;
    const int wave = tid >> 5;
    const int lane = tid & 31;
    const int mi   = wave & 3;                   // 0..3  row strip (32 rows)
    const int ni   = wave >> 2;                  // 0..1  col strip (32 cols)
    const int mtile = blockIdx.x >> 4;           // 128 M tiles
    const int ntile = blockIdx.x & 15;           // 16  N tiles
    const int t0 = mtile * 128;
    const int n0 = ntile * 64;
    const int m  = lane & 15;
    const int kh = lane >> 4;

    const unsigned smemBase = (unsigned)(uintptr_t)smem;

    // async-stage one K-tile (A: 128x32 in 2 rounds, B: 64x32 in 1 round)
    auto issue_stage = [&](int buf, int k0) {
        const unsigned sbase = smemBase + (unsigned)buf * STAGE_BYTES;
#pragma unroll
        for (int q = 0; q < 2; ++q) {
            const int c = q * 256 + tid;         // 0..511
            const int row = c >> 2;              // 0..127
            const int cc  = (c & 3) * 8;         // 0,8,16,24
            async_load_b128(sbase + (unsigned)(row * LDT + cc) * 2,
                            (const void*)(Xbf + (size_t)(t0 + row) * DM + k0 + cc));
        }
        {
            const int row = tid >> 2;            // 0..63
            const int cc  = (tid & 3) * 8;
            async_load_b128(sbase + B_OFF_BYTES + (unsigned)(row * LDT + cc) * 2,
                            (const void*)(Wbf + (size_t)(n0 + row) * DM + k0 + cc));
        }
    };

    v8f acc00 = {}, acc01 = {}, acc10 = {}, acc11 = {};

    const int ra0 = (mi * 32 + m) * LDT;         // A frag rows (strip 0)
    const int ra1 = (mi * 32 + 16 + m) * LDT;    // A frag rows (strip 1)
    const int rb0 = (ni * 32 + m) * LDT;         // B frag cols (strip 0)
    const int rb1 = (ni * 32 + 16 + m) * LDT;    // B frag cols (strip 1)
    const int k8  = kh * 8;

    auto compute_stage = [&](int buf) {
        const __bf16* As = smem + buf * STAGE_ELEMS;
        const __bf16* Bs = As + 128 * LDT;
        union { uint4 u[2]; v16bf v; } a0, a1, b0, b1;
        a0.u[0] = *(const uint4*)&As[ra0 + k8];
        a0.u[1] = *(const uint4*)&As[ra0 + 16 + k8];
        a1.u[0] = *(const uint4*)&As[ra1 + k8];
        a1.u[1] = *(const uint4*)&As[ra1 + 16 + k8];
        b0.u[0] = *(const uint4*)&Bs[rb0 + k8];
        b0.u[1] = *(const uint4*)&Bs[rb0 + 16 + k8];
        b1.u[0] = *(const uint4*)&Bs[rb1 + k8];
        b1.u[1] = *(const uint4*)&Bs[rb1 + 16 + k8];
        acc00 = __builtin_amdgcn_wmma_f32_16x16x32_bf16(
            false, a0.v, false, b0.v, (short)0, acc00, false, false);
        acc01 = __builtin_amdgcn_wmma_f32_16x16x32_bf16(
            false, a0.v, false, b1.v, (short)0, acc01, false, false);
        acc10 = __builtin_amdgcn_wmma_f32_16x16x32_bf16(
            false, a1.v, false, b0.v, (short)0, acc10, false, false);
        acc11 = __builtin_amdgcn_wmma_f32_16x16x32_bf16(
            false, a1.v, false, b1.v, (short)0, acc11, false, false);
    };

    // software pipeline: prologue stage 0, then issue kt+1 while computing kt
    issue_stage(0, 0);
    for (int kt = 0; kt < 31; ++kt) {
        issue_stage((kt + 1) & 1, (kt + 1) << 5);
        WAIT_ASYNCCNT_3();                       // stage kt's 3 ops complete
        __syncthreads();                         // visible to all waves
        compute_stage(kt & 1);
        __syncthreads();                         // done reading buf before reuse
    }
    WAIT_ASYNCCNT_0();
    __syncthreads();
    compute_stage(1);                            // kt = 31 -> buf 1

    // Epilogue: bias + softplus, reduce this wave's 32 columns, atomicAdd.
    // Acc layout: lane column n = lane&15; element r -> row r + 8*kh.
    const float bias0 = b_delta[n0 + ni * 32 + m];
    const float bias1 = b_delta[n0 + ni * 32 + 16 + m];
    float rs0[8], rs1[8];
#pragma unroll
    for (int r = 0; r < 8; ++r) {
        rs0[r] = softplusf(acc00[r] + bias0) + softplusf(acc01[r] + bias1);
        rs1[r] = softplusf(acc10[r] + bias0) + softplusf(acc11[r] + bias1);
    }
#pragma unroll
    for (int r = 0; r < 8; ++r) {
        rs0[r] += __shfl_xor(rs0[r], 1, 32);
        rs0[r] += __shfl_xor(rs0[r], 2, 32);
        rs0[r] += __shfl_xor(rs0[r], 4, 32);
        rs0[r] += __shfl_xor(rs0[r], 8, 32);
        rs1[r] += __shfl_xor(rs1[r], 1, 32);
        rs1[r] += __shfl_xor(rs1[r], 2, 32);
        rs1[r] += __shfl_xor(rs1[r], 4, 32);
        rs1[r] += __shfl_xor(rs1[r], 8, 32);
    }
    if (m == 0) {
        const int rbase = t0 + mi * 32 + kh * 8;
#pragma unroll
        for (int r = 0; r < 8; ++r) {
            atomicAdd(&delta_sum[rbase + r], rs0[r]);
            atomicAdd(&delta_sum[rbase + 16 + r], rs1[r]);
        }
    }
}

// ---------------------------------------------------------------------------
// Bx[t,n] = sum_d x[t,d] * B[n,d]     (block per token; 16 n x 16 j threads)
// ---------------------------------------------------------------------------
__global__ void __launch_bounds__(256)
bx_kernel(const float* __restrict__ x, const float* __restrict__ Bm,
          float* __restrict__ Bx) {
    const int t = blockIdx.x;
    const int n = threadIdx.x >> 4;
    const int j = threadIdx.x & 15;
    const float* xr = x + (size_t)t * DM;
    const float* br = Bm + (size_t)n * DM;
    float acc = 0.f;
#pragma unroll 8
    for (int i = 0; i < 64; ++i) {
        const int d = j + (i << 4);
        acc += xr[d] * br[d];
    }
    acc += __shfl_xor(acc, 1, 32);
    acc += __shfl_xor(acc, 2, 32);
    acc += __shfl_xor(acc, 4, 32);
    acc += __shfl_xor(acc, 8, 32);
    if (j == 0) Bx[(size_t)t * ST + n] = acc;
}

// ---------------------------------------------------------------------------
// Chunked parallel scan of h = dA*h + dBx along L (associative composition).
// 64 blocks = (b,n); 256 threads x 16 steps each; Hillis-Steele over composites.
// ---------------------------------------------------------------------------
__global__ void __launch_bounds__(256)
scan_kernel(const float* __restrict__ delta_sum, const float* __restrict__ Bx,
            const float* __restrict__ A_log, float* __restrict__ hs) {
    const int b = blockIdx.x >> 4;
    const int n = blockIdx.x & 15;
    const int tid = threadIdx.x;
    const int base = b * SEQLEN + tid * 16;
    const float An = -__expf(A_log[n]);

    float dA[16], dB[16];
    float a_loc = 1.f, u_loc = 0.f;
#pragma unroll
    for (int s = 0; s < 16; ++s) {
        const float delta = delta_sum[base + s] * (1.0f / (float)DM);
        const float da = __expf(delta * An);
        const float db = delta * Bx[(size_t)(base + s) * ST + n];
        dA[s] = da; dB[s] = db;
        a_loc = da * a_loc;
        u_loc = da * u_loc + db;
    }

    __shared__ float sA[256], sU[256];
    sA[tid] = a_loc; sU[tid] = u_loc;
    __syncthreads();
#pragma unroll
    for (int off = 1; off < 256; off <<= 1) {
        const float am = sA[tid], um = sU[tid];
        const float pa = (tid >= off) ? sA[tid - off] : 1.f;
        const float pu = (tid >= off) ? sU[tid - off] : 0.f;
        __syncthreads();
        sA[tid] = am * pa;
        sU[tid] = am * pu + um;
        __syncthreads();
    }

    float h = (tid == 0) ? 0.f : sU[tid - 1];    // exclusive prefix (h0 = 0)
#pragma unroll
    for (int s = 0; s < 16; ++s) {
        h = dA[s] * h + dB[s];
        hs[(size_t)(base + s) * ST + n] = h;
    }
}

// ---------------------------------------------------------------------------
// y[t,d] = sum_n hs[t,n]*C[d,n] + D[d]*x[t,d]      (block per token)
// ---------------------------------------------------------------------------
__global__ void __launch_bounds__(256)
out_kernel(const float* __restrict__ x, const float* __restrict__ hs,
           const float* __restrict__ Cm, const float* __restrict__ Dv,
           float* __restrict__ y) {
    const int t = blockIdx.x;
    __shared__ float sh[ST];
    if (threadIdx.x < ST) sh[threadIdx.x] = hs[(size_t)t * ST + threadIdx.x];
    __syncthreads();
    for (int d = threadIdx.x; d < DM; d += 256) {
        const float* c = Cm + (size_t)d * ST;
        float acc = Dv[d] * x[(size_t)t * DM + d];
#pragma unroll
        for (int k = 0; k < ST; ++k) acc += sh[k] * c[k];
        y[(size_t)t * DM + d] = acc;
    }
}

// ---------------------------------------------------------------------------
// host launch
// ---------------------------------------------------------------------------
extern "C" void kernel_launch(void* const* d_in, const int* in_sizes, int n_in,
                              void* d_out, int out_size, void* d_ws, size_t ws_size,
                              hipStream_t stream) {
    (void)in_sizes; (void)n_in; (void)out_size;
    const float* x     = (const float*)d_in[0];
    const float* A_log = (const float*)d_in[1];
    const float* Bm    = (const float*)d_in[2];
    const float* Cm    = (const float*)d_in[3];
    const float* Dv    = (const float*)d_in[4];
    const float* Wd    = (const float*)d_in[5];
    const float* b_del = (const float*)d_in[6];
    float* y = (float*)d_out;

    char* ws = (char*)d_ws;
    const size_t offWbf   = 0;                                  // 2 MB
    const size_t offDelta = offWbf + (size_t)DM * DM * 2;       // 64 KB
    const size_t offBx    = offDelta + (size_t)NTOK * 4;        // 1 MB
    const size_t offHs    = offBx + (size_t)NTOK * ST * 4;      // 1 MB
    const size_t offXbf   = offHs + (size_t)NTOK * ST * 4;      // 33.5 MB
    const size_t needAll  = offXbf + (size_t)NTOK * DM * 2;

    __bf16* Wbf      = (__bf16*)(ws + offWbf);
    float* delta_sum = (float*)(ws + offDelta);
    float* Bx        = (float*)(ws + offBx);
    float* hs        = (float*)(ws + offHs);
    // Prefer workspace for bf16 x; fall back to staging inside d_out
    // (d_out is 64 MB fp32, only written by out_kernel at the very end,
    // strictly after delta_gemm has consumed Xbf).
    __bf16* Xbf = (ws_size >= needAll) ? (__bf16*)(ws + offXbf) : (__bf16*)d_out;

    cvt_bf16_kernel<<<2048, 256, 0, stream>>>(x, Xbf, NTOK * DM / 8);
    cvt_bf16_kernel<<<512, 256, 0, stream>>>(Wd, Wbf, DM * DM / 8);

    hipMemsetAsync(delta_sum, 0, (size_t)NTOK * sizeof(float), stream);

    delta_gemm_kernel<<<dim3(2048), 256, 0, stream>>>(Xbf, Wbf, b_del, delta_sum);

    bx_kernel<<<NTOK, 256, 0, stream>>>(x, Bm, Bx);

    scan_kernel<<<NBATCH * ST, 256, 0, stream>>>(delta_sum, Bx, A_log, hs);

    out_kernel<<<NTOK, 256, 0, stream>>>(x, hs, Cm, Dv, y);
}
